// GATLayer_45140106281683
// MI455X (gfx1250) — matrix operand order repaired and checked by reading, not compiled, not run on hardware
//
#include <hip/hip_runtime.h>

typedef __attribute__((ext_vector_type(16))) _Float16 v16h;
typedef __attribute__((ext_vector_type(8)))  float    v8f;

#define Bc   8
#define Nn   4096
#define FIN  256
#define FOUT 64

// Workspace layout:
//   [0, 32768)              : WT, f16, [FOUT][FIN]  (W transposed + converted)
//   [32768, 32768 + 8.4MB)  : H,  f32, [B][N][FOUT] (stage-1 result)

// ---------------------------------------------------------------------------
// Kernel 0: W [FIN][FOUT] f32  ->  WT [FOUT][FIN] f16
// Tiny (16K elements); transposed layout makes each WMMA B-fragment a single
// contiguous 32-byte read per lane in stage 1.
// ---------------------------------------------------------------------------
__global__ void k_convert_W(const float* __restrict__ W, _Float16* __restrict__ WT) {
    for (int i = threadIdx.x; i < FOUT * FIN; i += blockDim.x) {
        int o = i >> 8;          // i / FIN
        int k = i & (FIN - 1);   // i % FIN
        WT[i] = (_Float16)W[k * FOUT + o];
    }
}

// ---------------------------------------------------------------------------
// Kernel 1: H = input_h @ W via v_wmma_f32_16x16x32_f16.
// One wave per 16-row M-strip of one batch; computes all 4 N-tiles (FOUT=64).
// A: 16x32 f16 tile of input_h (converted in-register from f32)
// B: 32x16 f16 tile of W, read from WT so each lane reads 16 contiguous halves
// C: 4 x v8f accumulators (f32)
// ---------------------------------------------------------------------------
__global__ __launch_bounds__(32) void k_stage1(const float* __restrict__ X,
                                               const _Float16* __restrict__ WT,
                                               float* __restrict__ H) {
    const int lid   = threadIdx.x;        // 0..31
    const int b     = blockIdx.x >> 8;    // 256 M-tiles per batch
    const int mtile = blockIdx.x & 255;
    const int m     = lid & 15;           // row within tile
    const int kh    = lid >> 4;           // lane half-group (ISA A/B layout)
    const int col   = lid & 15;           // B/C column

    const float* xrow = X + (size_t)(b * Nn + mtile * 16 + m) * FIN;

    v8f c0 = {}, c1 = {}, c2 = {}, c3 = {};

    for (int kk = 0; kk < FIN; kk += 32) {
        // ---- A fragment: halves j = K = kh*16 + j (contiguous, matches B)
        const float4* p0 = (const float4*)(xrow + kk + kh * 16);
        const float4* p1 = (const float4*)(xrow + kk + kh * 16 + 8);
        float4 f0 = p0[0], f1 = p0[1];
        float4 f2 = p1[0], f3 = p1[1];
        v16h a;
        a[0]  = (_Float16)f0.x; a[1]  = (_Float16)f0.y;
        a[2]  = (_Float16)f0.z; a[3]  = (_Float16)f0.w;
        a[4]  = (_Float16)f1.x; a[5]  = (_Float16)f1.y;
        a[6]  = (_Float16)f1.z; a[7]  = (_Float16)f1.w;
        a[8]  = (_Float16)f2.x; a[9]  = (_Float16)f2.y;
        a[10] = (_Float16)f2.z; a[11] = (_Float16)f2.w;
        a[12] = (_Float16)f3.x; a[13] = (_Float16)f3.y;
        a[14] = (_Float16)f3.z; a[15] = (_Float16)f3.w;

        // ---- B fragments: halves j = K = kh*16 + j, contiguous in WT[col][:]
        const _Float16* wt = WT + kk + kh * 16;
        v16h b0 = *(const v16h*)(wt + (size_t)(col     ) * FIN);
        v16h b1 = *(const v16h*)(wt + (size_t)(col + 16) * FIN);
        v16h b2 = *(const v16h*)(wt + (size_t)(col + 32) * FIN);
        v16h b3 = *(const v16h*)(wt + (size_t)(col + 48) * FIN);

        c0 = __builtin_amdgcn_wmma_f32_16x16x32_f16(false, a, false, b0, (short)0, c0, false, false);
        c1 = __builtin_amdgcn_wmma_f32_16x16x32_f16(false, a, false, b1, (short)0, c1, false, false);
        c2 = __builtin_amdgcn_wmma_f32_16x16x32_f16(false, a, false, b2, (short)0, c2, false, false);
        c3 = __builtin_amdgcn_wmma_f32_16x16x32_f16(false, a, false, b3, (short)0, c3, false, false);
    }

    // C/D layout: VGPR r -> M = r + kh*8 ; lanes' N = col
    float* hbase = H + (size_t)(b * Nn + mtile * 16) * FOUT;
    for (int r = 0; r < 8; ++r) {
        float* hp = hbase + (size_t)(r + kh * 8) * FOUT + col;
        hp[0]  = c0[r];
        hp[16] = c1[r];
        hp[32] = c2[r];
        hp[48] = c3[r];
    }
}

// ---------------------------------------------------------------------------
// Kernel 2: out[b][m][:] = sum over nonzero columns c of adj[m] of H[b][c][:]
// One block per row m. Phase 1: deterministic sorted compaction of nonzero
// columns into LDS (prefix scan, no atomics -> reproducible sum order).
// Phase 2: 8 waves (wave = batch) gather ~82 rows of H from L2 and accumulate.
// ---------------------------------------------------------------------------
__global__ __launch_bounds__(256) void k_stage2(const float* __restrict__ adj,
                                                const float* __restrict__ H,
                                                float* __restrict__ out) {
    __shared__ unsigned short s_list[Nn];
    __shared__ int s_cnt[256];
    __shared__ int s_off[256];
    __shared__ int s_total;

    const int m = blockIdx.x;
    const int t = threadIdx.x;

    // ---- Phase 1: scan adj row m, each thread owns a contiguous 16-col segment
    const float* arow = adj + (size_t)m * Nn + t * 16;
    float v[16];
    *(float4*)(&v[0])  = ((const float4*)arow)[0];
    *(float4*)(&v[4])  = ((const float4*)arow)[1];
    *(float4*)(&v[8])  = ((const float4*)arow)[2];
    *(float4*)(&v[12]) = ((const float4*)arow)[3];
    unsigned mask = 0;
    #pragma unroll
    for (int j = 0; j < 16; ++j)
        mask |= (v[j] != 0.0f ? 1u : 0u) << j;
    s_cnt[t] = __popc(mask);
    __syncthreads();

    if (t == 0) {   // serial prefix over 256 counts: trivial vs. 16KB row scan
        int run = 0;
        for (int i = 0; i < 256; ++i) { s_off[i] = run; run += s_cnt[i]; }
        s_total = run;
    }
    __syncthreads();

    {
        int off = s_off[t];
        #pragma unroll
        for (int j = 0; j < 16; ++j)
            if (mask & (1u << j)) s_list[off++] = (unsigned short)(t * 16 + j);
    }
    __syncthreads();

    // ---- Phase 2: wave = batch; lane covers features 2*lane, 2*lane+1
    const int b    = t >> 5;
    const int lane = t & 31;
    const float* hb = H + (size_t)b * Nn * FOUT;
    const int total = s_total;

    float accx = 0.0f, accy = 0.0f;
    for (int i = 0; i < total; ++i) {
        const int c = s_list[i];                       // uniform across wave
        const float2 hv = *(const float2*)(hb + (size_t)c * FOUT + lane * 2);
        accx += hv.x;
        accy += hv.y;
    }

    float2 r; r.x = accx; r.y = accy;
    *(float2*)(out + (size_t)(b * Nn + m) * FOUT + lane * 2) = r;
}

// ---------------------------------------------------------------------------
extern "C" void kernel_launch(void* const* d_in, const int* in_sizes, int n_in,
                              void* d_out, int out_size, void* d_ws, size_t ws_size,
                              hipStream_t stream) {
    const float* input_h = (const float*)d_in[0];   // [8,4096,256] f32
    const float* adj     = (const float*)d_in[1];   // [4096,4096]  f32 (0/1)
    const float* W       = (const float*)d_in[2];   // [256,64]     f32
    float* out = (float*)d_out;                     // [8,4096,64]  f32

    _Float16* WT = (_Float16*)d_ws;                       // 32 KB
    float*    H  = (float*)((char*)d_ws + 32768);         // 8.39 MB

    k_convert_W<<<1, 256, 0, stream>>>(W, WT);
    k_stage1<<<Bc * (Nn / 16), 32, 0, stream>>>(input_h, WT, H);
    k_stage2<<<Nn, 256, 0, stream>>>(adj, H, out);
}
